// GCN_62457414418905
// MI455X (gfx1250) — matrix operand order repaired and checked by reading, not compiled, not run on hardware
//
#include <hip/hip_runtime.h>

// ---------------- constants (match reference) ----------------
#define EMB      300
#define EMBP     320     // K padded to 10 * 32
#define NPADW    320     // layer-GEMM output cols padded to 5 * 64 (per-wave N-block)
#define NLAYER   5
#define NGRAPH   4096
#define FEAT     256
#define HID      128
#define NBOND    5
#define NDIR     3

typedef __bf16  bf16_t;
typedef bf16_t  v16bf __attribute__((ext_vector_type(16)));
typedef float   v8f   __attribute__((ext_vector_type(8)));

union Frag {
    v16bf v;
    uint4 q[2];
};

__device__ __forceinline__ unsigned short f2bf(float f) {
    unsigned int u = __float_as_uint(f);
    unsigned int r = u + 0x7FFFu + ((u >> 16) & 1u);   // round-to-nearest-even
    return (unsigned short)(r >> 16);
}
__device__ __forceinline__ float bf2f(unsigned short h) {
    return __uint_as_float(((unsigned int)h) << 16);
}

// ---------------- register-blocked bf16 WMMA GEMM ----------------
// C[M x Ncols] = A[M x K] * Bt[Ncols x K]^T (+ bias), K = Ksteps*32.
// A: bf16 row-major (lda elems), Bt[n][k]: bf16 (ldb elems).
// One wave computes a 32x64 block = 2x4 WMMA tiles (A frags reused 4x, B frags 2x).
// M must be a multiple of 32, Ncols a multiple of 64. Writes f32 (Cf) or bf16 (Cb).
__global__ void wmma_gemm_bf16(const unsigned short* __restrict__ A, int lda,
                               const unsigned short* __restrict__ Bt, int ldb,
                               float* __restrict__ Cf, unsigned short* __restrict__ Cb,
                               int ldc, int Mblk, int Nblk, int Ksteps,
                               const float* __restrict__ bias)
{
    int wave = blockIdx.x * (blockDim.x >> 5) + (threadIdx.x >> 5);
    if (wave >= Mblk * Nblk) return;              // wave-uniform; EXEC stays all-1s
    int mb = wave / Nblk;
    int nb = wave - mb * Nblk;
    int lane = threadIdx.x & 31;
    int lh = lane & 15;
    int hi = lane >> 4;

    // A frag (16-bit 16x32): lane lh holds row; k chunks {hi*8.., 16+hi*8..}
    const unsigned short* arow[2];
    arow[0] = A + (size_t)(mb * 32 + lh) * lda + hi * 8;
    arow[1] = arow[0] + (size_t)16 * lda;
    // B frag (16-bit 32x16): lane lh holds col; k chunks {hi*8.., 16+hi*8..}
    const unsigned short* brow[4];
    brow[0] = Bt + (size_t)(nb * 64 + lh) * ldb + hi * 8;
    brow[1] = brow[0] + (size_t)16 * ldb;
    brow[2] = brow[0] + (size_t)32 * ldb;
    brow[3] = brow[0] + (size_t)48 * ldb;

    v8f acc[2][4];
    #pragma unroll
    for (int i = 0; i < 2; ++i)
        #pragma unroll
        for (int j = 0; j < 4; ++j)
            acc[i][j] = {};

    for (int ks = 0; ks < Ksteps; ++ks) {
        Frag a[2], b[4];
        #pragma unroll
        for (int i = 0; i < 2; ++i) {
            const unsigned short* ap = arow[i] + ks * 32;
            a[i].q[0] = *(const uint4*)(ap);        // k base + hi*8      .. +7
            a[i].q[1] = *(const uint4*)(ap + 16);   // k base + 16 + hi*8 .. +7
        }
        #pragma unroll
        for (int j = 0; j < 4; ++j) {
            const unsigned short* bp = brow[j] + ks * 32;
            b[j].q[0] = *(const uint4*)(bp);        // k base + hi*8      .. +7
            b[j].q[1] = *(const uint4*)(bp + 16);   // k base + 16 + hi*8 .. +7
        }
        #pragma unroll
        for (int i = 0; i < 2; ++i)
            #pragma unroll
            for (int j = 0; j < 4; ++j)
                acc[i][j] = __builtin_amdgcn_wmma_f32_16x16x32_bf16(
                                false, a[i].v, false, b[j].v, (short)0, acc[i][j],
                                false, false);
    }

    // C layout per 16x16 tile: VGPR r -> (M = hi*8 + r, N = lh)
    #pragma unroll
    for (int j = 0; j < 4; ++j) {
        int ccol = nb * 64 + j * 16 + lh;
        float bv = bias ? bias[ccol] : 0.f;
        #pragma unroll
        for (int i = 0; i < 2; ++i) {
            int crow = mb * 32 + i * 16 + hi * 8;
            if (Cf) {
                #pragma unroll
                for (int r = 0; r < 8; ++r)
                    Cf[(size_t)(crow + r) * ldc + ccol] = acc[i][j][r] + bv;
            } else {
                #pragma unroll
                for (int r = 0; r < 8; ++r)
                    Cb[(size_t)(crow + r) * ldc + ccol] = f2bf(acc[i][j][r] + bv);
            }
        }
    }
}

// ---------------- pack W (K x Ncol, f32) -> Wt (Npad x Kpad, bf16), zero-padded ----
__global__ void pack_t_kernel(const float* __restrict__ src, unsigned short* __restrict__ dst,
                              int K, int Ncol, int Kpad, int Npad)
{
    int t = blockIdx.x * blockDim.x + threadIdx.x;
    if (t >= Npad * Kpad) return;
    int n = t / Kpad, k = t - n * Kpad;
    float v = (k < K && n < Ncol) ? src[(size_t)k * Ncol + n] : 0.f;
    dst[(size_t)n * Kpad + k] = f2bf(v);
}

// ---------------- h = emb1[atom] + emb2[chi], bf16, K- and M-padded ----------------
__global__ void embed_kernel(const int* __restrict__ at, const int* __restrict__ ch,
                             const float* __restrict__ e1, const float* __restrict__ e2,
                             unsigned short* __restrict__ h, int N, int Mpad)
{
    int t = blockIdx.x * blockDim.x + threadIdx.x;
    if (t >= Mpad * EMBP) return;
    int n = t / EMBP, c = t - n * EMBP;
    float v = 0.f;
    if (n < N && c < EMB) v = e1[(size_t)at[n] * EMB + c] + e2[(size_t)ch[n] * EMB + c];
    h[(size_t)n * EMBP + c] = f2bf(v);
}

// ---------------- out = b[l] (broadcast), padding zeroed ----------------
__global__ void out_init_kernel(float* __restrict__ out, const float* __restrict__ bl, int N)
{
    int t = blockIdx.x * blockDim.x + threadIdx.x;
    if (t >= N * EMBP) return;
    int c = t % EMBP;
    out[t] = (c < EMB) ? bl[c] : 0.f;
}

// ---------------- edge message + segment-sum scatter ----------------
// one block per edge (incl. self-loops appended), 320 threads, c<300 active
__global__ void scatter_kernel(const unsigned short* __restrict__ hx,
                               const int* __restrict__ ei, const int* __restrict__ et,
                               const int* __restrict__ ed,
                               const float* __restrict__ ee1l, const float* __restrict__ ee2l,
                               float* __restrict__ out, int E)
{
    int e = blockIdx.x;
    int c = threadIdx.x;
    if (c >= EMB) return;
    int row, col, bt, dr;
    if (e < E) { row = ei[e]; col = ei[E + e]; bt = et[e]; dr = ed[e]; }
    else       { row = e - E; col = e - E;     bt = 4;     dr = 0;     }
    float ee = ee1l[bt] + ee2l[dr];
    float v = bf2f(hx[(size_t)row * EMBP + c]) + ee;
    atomicAdd(&out[(size_t)col * EMBP + c], v);
}

// ---------------- BatchNorm: stats ----------------
__global__ void stats_zero_kernel(float* __restrict__ sums, float* __restrict__ sumsq)
{
    int c = threadIdx.x;
    if (c < EMBP) { sums[c] = 0.f; sumsq[c] = 0.f; }
}

__global__ void stats_kernel(const float* __restrict__ out,
                             float* __restrict__ sums, float* __restrict__ sumsq, int N)
{
    int c = threadIdx.x;                 // 320 threads, coalesced over columns
    if (c >= EMBP) return;
    int r0 = blockIdx.x * 256;
    int r1 = min(r0 + 256, N);
    float s = 0.f, s2 = 0.f;
    for (int r = r0; r < r1; ++r) {
        float v = out[(size_t)r * EMBP + c];
        s += v; s2 += v * v;
    }
    atomicAdd(&sums[c], s);
    atomicAdd(&sumsq[c], s2);
}

__global__ void stats_final_kernel(const float* __restrict__ sums, const float* __restrict__ sumsq,
                                   const float* __restrict__ gamma, const float* __restrict__ beta,
                                   float* __restrict__ scale, float* __restrict__ sbias, int N)
{
    int c = threadIdx.x;
    if (c >= EMBP) return;
    float invN = 1.f / (float)N;
    float mean = sums[c] * invN;
    float var  = sumsq[c] * invN - mean * mean;
    float sc = 0.f, bs = 0.f;
    if (c < EMB) {
        sc = gamma[c] * rsqrtf(var + 1e-5f);
        bs = beta[c] - mean * sc;
    }
    scale[c] = sc;
    sbias[c] = bs;
}

// ---------------- BN apply (+ReLU), repack bf16; last layer keeps f32 in place ----
__global__ void bn_apply_kernel(float* __restrict__ out,
                                const float* __restrict__ scale, const float* __restrict__ sbias,
                                unsigned short* __restrict__ h, int N, int do_relu, int write_f32)
{
    int t = blockIdx.x * blockDim.x + threadIdx.x;
    if (t >= N * EMBP) return;
    int c = t % EMBP;
    float v = out[t] * scale[c] + sbias[c];
    if (do_relu) v = fmaxf(v, 0.f);
    h[t] = f2bf(v);
    if (write_f32) out[t] = v;
}

// ---------------- global mean pool ----------------
__global__ void pool_zero_kernel(float* __restrict__ hg, float* __restrict__ cnt)
{
    int t = blockIdx.x * blockDim.x + threadIdx.x;
    if (t < NGRAPH * EMBP) hg[t] = 0.f;
    if (t < NGRAPH) cnt[t] = 0.f;
}

__global__ void pool_scatter_kernel(const float* __restrict__ hlast, const int* __restrict__ batch,
                                    float* __restrict__ hg, float* __restrict__ cnt)
{
    int n = blockIdx.x;
    int c = threadIdx.x;
    if (c >= EMBP) return;
    int g = batch[n];
    atomicAdd(&hg[(size_t)g * EMBP + c], hlast[(size_t)n * EMBP + c]);
    if (c == 0) atomicAdd(&cnt[g], 1.f);
}

__global__ void pool_div_kernel(const float* __restrict__ hg, const float* __restrict__ cnt,
                                unsigned short* __restrict__ hgb)
{
    int t = blockIdx.x * blockDim.x + threadIdx.x;
    if (t >= NGRAPH * EMBP) return;
    int g = t / EMBP;
    float ct = fmaxf(cnt[g], 1.f);
    hgb[t] = f2bf(hg[t] / ct);
}

// ---------------- misc elementwise ----------------
__global__ void f32_to_bf16_kernel(const float* __restrict__ src, unsigned short* __restrict__ dst, int n)
{
    int t = blockIdx.x * blockDim.x + threadIdx.x;
    if (t < n) dst[t] = f2bf(src[t]);
}

__global__ void softplus_kernel(float* __restrict__ x, int n)
{
    int t = blockIdx.x * blockDim.x + threadIdx.x;
    if (t >= n) return;
    float v = x[t];
    x[t] = (v > 20.f) ? v : log1pf(expf(v));
}

// ---------------- tiny head: pred = hm @ W2 + b2  (128 -> 2) ----------------
__global__ void head2_kernel(const float* __restrict__ hm, const float* __restrict__ W2,
                             const float* __restrict__ b2, float* __restrict__ pred)
{
    int t = blockIdx.x * blockDim.x + threadIdx.x;
    if (t >= NGRAPH * 2) return;
    int g = t >> 1, j = t & 1;
    float s = b2[j];
    const float* hr = hm + (size_t)g * HID;
    for (int k = 0; k < HID; ++k) s += hr[k] * W2[k * 2 + j];
    pred[t] = s;
}

// ---------------- host orchestration ----------------
extern "C" void kernel_launch(void* const* d_in, const int* in_sizes, int n_in,
                              void* d_out, int out_size, void* d_ws, size_t ws_size,
                              hipStream_t stream)
{
    const int*   atom_type = (const int*)  d_in[0];
    const int*   chirality = (const int*)  d_in[1];
    const int*   edge_idx  = (const int*)  d_in[2];
    const int*   edge_type = (const int*)  d_in[3];
    const int*   edge_dir  = (const int*)  d_in[4];
    const int*   batch     = (const int*)  d_in[5];
    const float* emb1      = (const float*)d_in[6];
    const float* emb2      = (const float*)d_in[7];
    const float* W         = (const float*)d_in[8];
    const float* bvec      = (const float*)d_in[9];
    const float* ee1       = (const float*)d_in[10];
    const float* ee2       = (const float*)d_in[11];
    const float* bn_gamma  = (const float*)d_in[12];
    const float* bn_beta   = (const float*)d_in[13];
    const float* feat_W    = (const float*)d_in[14];
    const float* feat_b    = (const float*)d_in[15];
    const float* head_W1   = (const float*)d_in[16];
    const float* head_b1   = (const float*)d_in[17];
    const float* head_W2   = (const float*)d_in[18];
    const float* head_b2   = (const float*)d_in[19];

    const int N    = in_sizes[0];                 // 100000
    const int E    = in_sizes[3];                 // 200000
    const int ET   = E + N;                       // edges incl. self loops
    const int Mpad = ((N + 31) / 32) * 32;        // M rounded up for 32-row wave blocks

    // ---- workspace carve-up (256B aligned regions) ----
    char* ws = (char*)d_ws;
    size_t off = 0;
    auto carve = [&](size_t bytes) -> char* {
        char* p = ws + off;
        off += (bytes + 255) & ~(size_t)255;
        return p;
    };
    unsigned short* h_bf    = (unsigned short*)carve((size_t)Mpad * EMBP * 2);
    unsigned short* hx_bf   = (unsigned short*)carve((size_t)Mpad * EMBP * 2);
    float*          outb    = (float*)         carve((size_t)N * EMBP * 4);
    unsigned short* Wt      = (unsigned short*)carve((size_t)NLAYER * NPADW * EMBP * 2);
    unsigned short* featWt  = (unsigned short*)carve((size_t)FEAT * EMBP * 2);
    unsigned short* hW1t    = (unsigned short*)carve((size_t)HID * FEAT * 2);
    float*          sums    = (float*)         carve(EMBP * 4);
    float*          sumsq   = (float*)         carve(EMBP * 4);
    float*          scale   = (float*)         carve(EMBP * 4);
    float*          sbias   = (float*)         carve(EMBP * 4);
    float*          hg_sum  = (float*)         carve((size_t)NGRAPH * EMBP * 4);
    float*          hg_cnt  = (float*)         carve((size_t)NGRAPH * 4);
    unsigned short* hg_bf   = (unsigned short*)carve((size_t)NGRAPH * EMBP * 2);
    unsigned short* hf_bf   = (unsigned short*)carve((size_t)NGRAPH * FEAT * 2);
    float*          hm      = (float*)         carve((size_t)NGRAPH * HID * 4);
    (void)ws_size; (void)n_in; (void)out_size;

    float* hf_out   = (float*)d_out;                         // [4096 x 256]
    float* pred_out = (float*)d_out + (size_t)NGRAPH * FEAT; // [4096 x 2]

    const int TB = 256;
    auto blocks = [](long long n, int tb) { return (int)((n + tb - 1) / tb); };

    // ---- pack weights (bf16, transposed, zero-padded) ----
    for (int l = 0; l < NLAYER; ++l)
        pack_t_kernel<<<blocks((long long)NPADW * EMBP, TB), TB, 0, stream>>>(
            W + (size_t)l * EMB * EMB, Wt + (size_t)l * NPADW * EMBP, EMB, EMB, EMBP, NPADW);
    pack_t_kernel<<<blocks((long long)FEAT * EMBP, TB), TB, 0, stream>>>(
        feat_W, featWt, EMB, FEAT, EMBP, FEAT);
    pack_t_kernel<<<blocks((long long)HID * FEAT, TB), TB, 0, stream>>>(
        head_W1, hW1t, FEAT, HID, FEAT, HID);

    // ---- initial embedding (zero-fills M padding rows) ----
    embed_kernel<<<blocks((long long)Mpad * EMBP, TB), TB, 0, stream>>>(
        atom_type, chirality, emb1, emb2, h_bf, N, Mpad);

    // ---- 5 GCN layers ----
    const int MblkL = Mpad / 32;     // 3125
    const int NblkL = NPADW / 64;    // 5
    for (int l = 0; l < NLAYER; ++l) {
        // hx = h @ W[l]   (bf16 WMMA, 32x64 per wave, bf16 output)
        wmma_gemm_bf16<<<blocks(MblkL * NblkL, 8), 256, 0, stream>>>(
            h_bf, EMBP, Wt + (size_t)l * NPADW * EMBP, EMBP,
            nullptr, hx_bf, EMBP, MblkL, NblkL, EMBP / 32, nullptr);

        // out = b[l], then scatter-add messages
        out_init_kernel<<<blocks((long long)N * EMBP, TB), TB, 0, stream>>>(
            outb, bvec + (size_t)l * EMB, N);
        scatter_kernel<<<ET, EMBP, 0, stream>>>(
            hx_bf, edge_idx, edge_type, edge_dir,
            ee1 + (size_t)l * NBOND, ee2 + (size_t)l * NDIR, outb, E);

        // BatchNorm (training-mode batch stats)
        stats_zero_kernel<<<1, EMBP, 0, stream>>>(sums, sumsq);
        stats_kernel<<<blocks(N, 256), EMBP, 0, stream>>>(outb, sums, sumsq, N);
        stats_final_kernel<<<1, EMBP, 0, stream>>>(
            sums, sumsq, bn_gamma + (size_t)l * EMB, bn_beta + (size_t)l * EMB,
            scale, sbias, N);
        bn_apply_kernel<<<blocks((long long)N * EMBP, TB), TB, 0, stream>>>(
            outb, scale, sbias, h_bf, N, (l < NLAYER - 1) ? 1 : 0, (l == NLAYER - 1) ? 1 : 0);
    }

    // ---- global mean pool (last layer f32 kept in `outb`) ----
    pool_zero_kernel<<<blocks((long long)NGRAPH * EMBP, TB), TB, 0, stream>>>(hg_sum, hg_cnt);
    pool_scatter_kernel<<<N, EMBP, 0, stream>>>(outb, batch, hg_sum, hg_cnt);
    pool_div_kernel<<<blocks((long long)NGRAPH * EMBP, TB), TB, 0, stream>>>(hg_sum, hg_cnt, hg_bf);

    // ---- hf = hg @ feat_W + feat_b  -> d_out (f32) ----
    wmma_gemm_bf16<<<blocks((NGRAPH / 32) * (FEAT / 64), 8), 256, 0, stream>>>(
        hg_bf, EMBP, featWt, EMBP, hf_out, nullptr, FEAT,
        NGRAPH / 32, FEAT / 64, EMBP / 32, feat_b);

    // ---- head: hm = softplus(hf @ W1 + b1); pred = hm @ W2 + b2 ----
    f32_to_bf16_kernel<<<blocks((long long)NGRAPH * FEAT, TB), TB, 0, stream>>>(
        hf_out, hf_bf, NGRAPH * FEAT);
    wmma_gemm_bf16<<<blocks((NGRAPH / 32) * (HID / 64), 8), 256, 0, stream>>>(
        hf_bf, FEAT, hW1t, FEAT, hm, nullptr, HID,
        NGRAPH / 32, HID / 64, FEAT / 32, head_b1);
    softplus_kernel<<<blocks((long long)NGRAPH * HID, TB), TB, 0, stream>>>(hm, NGRAPH * HID);
    head2_kernel<<<blocks(NGRAPH * 2, TB), TB, 0, stream>>>(hm, head_W2, head_b2, pred_out);
}
